// EdgeNet_45810121179634
// MI455X (gfx1250) — matrix operand order repaired and checked by reading, not compiled, not run on hardware
//
#include <hip/hip_runtime.h>
#include <math.h>

// ---------------------------------------------------------------------------
// EdgeNet for MI455X (gfx1250, wave32).
// Edge MLPs run on V_WMMA_F32_16X16X32_F16 (f16 A/B, f32 accum), 16 edges per
// wave-tile, weights resident in VGPRs, msg rows staged in LDS, read as A fragments
// in contiguous k order (k = 16h + s). Aggregation via global_atomic_add_f32 (hits L2).
// ---------------------------------------------------------------------------

typedef __attribute__((ext_vector_type(16))) _Float16 v16h;
typedef __attribute__((ext_vector_type(8)))  _Float16 v8h;
typedef __attribute__((ext_vector_type(8)))  float    v8f;

#define CIN   71   // 2*(32+3)+1
#define KPAD  96   // 3 k-steps of 32
#define HID   64   // layer-1 width
#define HOUT  32   // conv layer-2 width

// ---- helpers ---------------------------------------------------------------

// Build a 16-bit A fragment (16x32 slice, this lane's view) from an LDS row.
// Low lanes (hi=0) take K = base+{0..15}; high lanes K = base+{16..31}.
__device__ __forceinline__ v16h a_frag(const _Float16* row, int base, int hi) {
  const v8h lo = *(const v8h*)(row + base + 16 * hi);
  const v8h hh = *(const v8h*)(row + base + 16 * hi + 8);
  v16h a;
#pragma unroll
  for (int i = 0; i < 8; ++i) { a[i] = lo[i]; a[8 + i] = hh[i]; }
  return a;
}

// Build a 16-bit B fragment from a row-major f32 weight matrix W[kRows x ldn].
// Lane n (low half) holds K = 32*ks+0..15 of column col; high half K += 16.
__device__ __forceinline__ v16h b_frag(const float* W, int ldn, int kRows,
                                       int kbase, int col) {
  v16h b;
#pragma unroll
  for (int i = 0; i < 16; ++i) {
    int K = kbase + i;
    float v = (K < kRows) ? W[K * ldn + col] : 0.0f;
    b[i] = (_Float16)v;
  }
  return b;
}

__device__ __forceinline__ v8f splat8(float x) {
  v8f c;
#pragma unroll
  for (int i = 0; i < 8; ++i) c[i] = x;
  return c;
}

#define WMMA_F16(A, B, C) \
  __builtin_amdgcn_wmma_f32_16x16x32_f16(false, (A), false, (B), (short)0, (C), false, false)

// ---- generic zero ----------------------------------------------------------

__global__ void zero_kernel(float* p, long n) {
  long i = (long)blockIdx.x * blockDim.x + threadIdx.x;
  long stride = (long)gridDim.x * blockDim.x;
  for (; i < n; i += stride) p[i] = 0.0f;
}

// ---- batchnorm stats: stats[0..2]=sum, stats[3..5]=sumsq --------------------

__global__ void bn_stats_kernel(const float* __restrict__ x, float* stats, int N) {
  __shared__ float red[256];
  float s[3] = {0.f, 0.f, 0.f}, q[3] = {0.f, 0.f, 0.f};
  for (int i = blockIdx.x * blockDim.x + threadIdx.x; i < N;
       i += gridDim.x * blockDim.x) {
#pragma unroll
    for (int c = 0; c < 3; ++c) {
      float v = x[i * 3 + c];
      s[c] += v;
      q[c] += v * v;
    }
  }
#pragma unroll
  for (int comp = 0; comp < 6; ++comp) {
    float v = (comp < 3) ? s[comp] : q[comp - 3];
    red[threadIdx.x] = v;
    __syncthreads();
    for (int off = 128; off > 0; off >>= 1) {
      if (threadIdx.x < off) red[threadIdx.x] += red[threadIdx.x + off];
      __syncthreads();
    }
    if (threadIdx.x == 0) atomicAdd(&stats[comp], red[0]);
    __syncthreads();
  }
}

// ---- node MLP: feat[n] = [tanh(relu(X@w1+b1)@w2+b2) (32), X (3), 0] --------

__global__ void node_kernel(const float* __restrict__ x,
                            const float* __restrict__ gamma,
                            const float* __restrict__ beta,
                            const float* __restrict__ w1, const float* __restrict__ b1,
                            const float* __restrict__ w2, const float* __restrict__ b2,
                            const float* __restrict__ stats,
                            float* __restrict__ feat, int N) {
  __shared__ float sw1[96], sb1[32], sw2[1024], sb2[32];
  for (int i = threadIdx.x; i < 96; i += blockDim.x) sw1[i] = w1[i];
  for (int i = threadIdx.x; i < 32; i += blockDim.x) { sb1[i] = b1[i]; sb2[i] = b2[i]; }
  for (int i = threadIdx.x; i < 1024; i += blockDim.x) sw2[i] = w2[i];
  __syncthreads();

  int n = blockIdx.x * blockDim.x + threadIdx.x;
  if (n >= N) return;

  float Ninv = 1.0f / (float)N;
  float X[3];
#pragma unroll
  for (int c = 0; c < 3; ++c) {
    float mu  = stats[c] * Ninv;
    float var = stats[3 + c] * Ninv - mu * mu;
    float rs  = rsqrtf(var + 1e-5f);
    X[c] = (x[n * 3 + c] - mu) * rs * gamma[c] + beta[c];
  }
  float h1[32];
#pragma unroll
  for (int j = 0; j < 32; ++j) {
    float a = X[0] * sw1[j] + X[1] * sw1[32 + j] + X[2] * sw1[64 + j] + sb1[j];
    h1[j] = fmaxf(a, 0.0f);
  }
  float* fr = feat + (long)n * 36;
#pragma unroll 4
  for (int j = 0; j < 32; ++j) {
    float acc = sb2[j];
#pragma unroll
    for (int k = 0; k < 32; ++k) acc += h1[k] * sw2[k * 32 + j];
    fr[j] = tanhf(acc);
  }
  fr[32] = X[0]; fr[33] = X[1]; fr[34] = X[2]; fr[35] = 0.0f;
}

// ---- feat2[n] = [Hn[n] (32), X[n] (3), 0] ----------------------------------

__global__ void feat2_kernel(const float* __restrict__ Hn,
                             const float* __restrict__ feat,
                             float* __restrict__ feat2, int N) {
  long i = (long)blockIdx.x * blockDim.x + threadIdx.x;
  if (i >= (long)N * 36) return;
  int n = (int)(i / 36), c = (int)(i % 36);
  float v;
  if (c < 32)      v = Hn[(long)n * 32 + c];
  else if (c < 35) v = feat[i];
  else             v = 0.0f;
  feat2[i] = v;
}

// ---- conv edge kernel: msg MLP + atomic scatter -----------------------------
// One wave (32 threads) per workgroup; 16 edges per tile.

__global__ void __launch_bounds__(32)
conv_kernel(const float* __restrict__ feat, const int* __restrict__ eidx,
            const float* __restrict__ eattr,
            const float* __restrict__ w_c1, const float* __restrict__ b_c1,
            const float* __restrict__ w_c2, const float* __restrict__ b_c2,
            float* __restrict__ Hn, int E, int numTiles) {
  __shared__ _Float16 msg[16 * KPAD];   // 16 rows x 96 (zero-padded K)
  __shared__ float    xi_s[16 * 36];    // f32 x_i rows for the diff
  __shared__ int      didx[16];         // dst index per row
  __shared__ _Float16 act[16 * HID];    // relu'd layer-1 activations

  const int lane = threadIdx.x;
  const int n    = lane & 15;
  const int hi   = lane >> 4;
  const int kb   = hi * 16;

  for (int i = lane; i < 16 * KPAD; i += 32) msg[i] = (_Float16)0.0f;

  // Resident weight fragments.
  v16h B1[4][3];
#pragma unroll
  for (int t = 0; t < 4; ++t)
#pragma unroll
    for (int ks = 0; ks < 3; ++ks)
      B1[t][ks] = b_frag(w_c1, HID, CIN, 32 * ks + kb, 16 * t + n);
  v16h B2[2][2];
#pragma unroll
  for (int t = 0; t < 2; ++t)
#pragma unroll
    for (int ks = 0; ks < 2; ++ks)
      B2[t][ks] = b_frag(w_c2, HOUT, HID, 32 * ks + kb, 16 * t + n);
  float bias1[4], bias2[2];
#pragma unroll
  for (int t = 0; t < 4; ++t) bias1[t] = b_c1[16 * t + n];
#pragma unroll
  for (int t = 0; t < 2; ++t) bias2[t] = b_c2[16 * t + n];

  const int* srcI = eidx;
  const int* dstI = eidx + E;

  for (int tile = blockIdx.x; tile < numTiles; tile += gridDim.x) {
    const int e0 = tile * 16;
    const int e  = e0 + n;
    const bool ev = (e < E);
    __syncthreads();

    if (hi == 0) {                       // x_i = feat[dst]
      int di = ev ? dstI[e] : -1;
      didx[n] = di;
      if (ev) {
        const float4* f4 = (const float4*)(feat + (long)di * 36);
        float xr[36];
#pragma unroll
        for (int q = 0; q < 9; ++q) {
          float4 v = f4[q];
          xr[4 * q] = v.x; xr[4 * q + 1] = v.y; xr[4 * q + 2] = v.z; xr[4 * q + 3] = v.w;
        }
#pragma unroll
        for (int c = 0; c < 35; ++c) {
          xi_s[n * 36 + c] = xr[c];
          msg[n * KPAD + c] = (_Float16)xr[c];
        }
      } else {
#pragma unroll
        for (int c = 0; c < 35; ++c) { xi_s[n * 36 + c] = 0.f; msg[n * KPAD + c] = (_Float16)0.f; }
      }
    }
    __syncthreads();
    if (hi == 1) {                       // x_j - x_i, attr
      if (ev) {
        int sj = srcI[e];
        const float4* f4 = (const float4*)(feat + (long)sj * 36);
        float xr[36];
#pragma unroll
        for (int q = 0; q < 9; ++q) {
          float4 v = f4[q];
          xr[4 * q] = v.x; xr[4 * q + 1] = v.y; xr[4 * q + 2] = v.z; xr[4 * q + 3] = v.w;
        }
#pragma unroll
        for (int c = 0; c < 35; ++c)
          msg[n * KPAD + 35 + c] = (_Float16)(xr[c] - xi_s[n * 36 + c]);
        msg[n * KPAD + 70] = (_Float16)eattr[e];
      } else {
#pragma unroll
        for (int c = 0; c < 36; ++c) msg[n * KPAD + 35 + c] = (_Float16)0.f;
      }
    }
    __syncthreads();

    // Layer 1: [16x96(f16)] @ [96x64] + b, relu
    const _Float16* row = msg + n * KPAD;
    v16h A0 = a_frag(row, 0,  hi);
    v16h A1 = a_frag(row, 32, hi);
    v16h A2 = a_frag(row, 64, hi);
    v8f C[4];
#pragma unroll
    for (int t = 0; t < 4; ++t) {
      C[t] = splat8(bias1[t]);
      C[t] = WMMA_F16(A0, B1[t][0], C[t]);
      C[t] = WMMA_F16(A1, B1[t][1], C[t]);
      C[t] = WMMA_F16(A2, B1[t][2], C[t]);
#pragma unroll
      for (int r = 0; r < 8; ++r) C[t][r] = fmaxf(C[t][r], 0.0f);
    }

    // Re-stage activations into A layout via LDS.
#pragma unroll
    for (int t = 0; t < 4; ++t)
#pragma unroll
      for (int r = 0; r < 8; ++r)
        act[(r + 8 * hi) * HID + 16 * t + n] = (_Float16)C[t][r];
    __syncthreads();

    // Layer 2: [16x64(f16)] @ [64x32] + b, tanh
    const _Float16* arow = act + n * HID;
    v16h A20 = a_frag(arow, 0,  hi);
    v16h A21 = a_frag(arow, 32, hi);
    v8f C2[2];
#pragma unroll
    for (int t = 0; t < 2; ++t) {
      C2[t] = splat8(bias2[t]);
      C2[t] = WMMA_F16(A20, B2[t][0], C2[t]);
      C2[t] = WMMA_F16(A21, B2[t][1], C2[t]);
#pragma unroll
      for (int r = 0; r < 8; ++r) C2[t][r] = tanhf(C2[t][r]);
    }

    // Scatter-add into Hn[dst].
#pragma unroll
    for (int t = 0; t < 2; ++t)
#pragma unroll
      for (int r = 0; r < 8; ++r) {
        int m  = r + 8 * hi;
        int di = didx[m];
        if (di >= 0) atomicAdd(&Hn[(long)di * 32 + 16 * t + n], C2[t][r]);
      }
  }
}

// ---- edge head: sigmoid(relu(e_in@w_e1+b)@w_e2+b) ---------------------------

__global__ void __launch_bounds__(32)
edgeout_kernel(const float* __restrict__ feat2, const int* __restrict__ eidx,
               const float* __restrict__ eattr,
               const float* __restrict__ w_e1, const float* __restrict__ b_e1,
               const float* __restrict__ w_e2, const float* __restrict__ b_e2,
               float* __restrict__ out, int E, int numTiles) {
  __shared__ _Float16 msg[16 * KPAD];

  const int lane = threadIdx.x;
  const int n    = lane & 15;
  const int hi   = lane >> 4;
  const int kb   = hi * 16;

  for (int i = lane; i < 16 * KPAD; i += 32) msg[i] = (_Float16)0.0f;

  v16h B1[4][3];
#pragma unroll
  for (int t = 0; t < 4; ++t)
#pragma unroll
    for (int ks = 0; ks < 3; ++ks)
      B1[t][ks] = b_frag(w_e1, HID, CIN, 32 * ks + kb, 16 * t + n);
  float bias1[4], w2l[4];
#pragma unroll
  for (int t = 0; t < 4; ++t) { bias1[t] = b_e1[16 * t + n]; w2l[t] = w_e2[16 * t + n]; }
  const float be2 = b_e2[0];

  const int* srcI = eidx;
  const int* dstI = eidx + E;

  for (int tile = blockIdx.x; tile < numTiles; tile += gridDim.x) {
    const int e0 = tile * 16;
    const int e  = e0 + n;
    const bool ev = (e < E);
    __syncthreads();

    {
      int idx = ev ? (hi ? dstI[e] : srcI[e]) : 0;
      const float4* f4 = (const float4*)(feat2 + (long)idx * 36);
      float xr[36];
#pragma unroll
      for (int q = 0; q < 9; ++q) {
        float4 v = f4[q];
        xr[4 * q] = v.x; xr[4 * q + 1] = v.y; xr[4 * q + 2] = v.z; xr[4 * q + 3] = v.w;
      }
      int base = hi ? 35 : 0;
#pragma unroll
      for (int c = 0; c < 35; ++c)
        msg[n * KPAD + base + c] = ev ? (_Float16)xr[c] : (_Float16)0.f;
      if (hi) msg[n * KPAD + 70] = ev ? (_Float16)eattr[e] : (_Float16)0.f;
    }
    __syncthreads();

    const _Float16* row = msg + n * KPAD;
    v16h A0 = a_frag(row, 0,  hi);
    v16h A1 = a_frag(row, 32, hi);
    v16h A2 = a_frag(row, 64, hi);
    v8f C[4];
#pragma unroll
    for (int t = 0; t < 4; ++t) {
      C[t] = splat8(bias1[t]);
      C[t] = WMMA_F16(A0, B1[t][0], C[t]);
      C[t] = WMMA_F16(A1, B1[t][1], C[t]);
      C[t] = WMMA_F16(A2, B1[t][2], C[t]);
#pragma unroll
      for (int r = 0; r < 8; ++r) C[t][r] = fmaxf(C[t][r], 0.0f);
    }

    // 64 -> 1 dot: per-lane partial (this lane holds columns 16t+n), then
    // xor-reduce within each 16-lane half (matches C-matrix N striping).
    float p[8];
#pragma unroll
    for (int r = 0; r < 8; ++r)
      p[r] = C[0][r] * w2l[0] + C[1][r] * w2l[1] + C[2][r] * w2l[2] + C[3][r] * w2l[3];
#pragma unroll
    for (int off = 8; off > 0; off >>= 1)
#pragma unroll
      for (int r = 0; r < 8; ++r) p[r] += __shfl_xor(p[r], off, 32);

    if (n == 0) {
#pragma unroll
      for (int r = 0; r < 8; ++r) {
        int eo = e0 + r + 8 * hi;
        if (eo < E) out[eo] = 1.0f / (1.0f + expf(-(p[r] + be2)));
      }
    }
  }
}

// ---- launch ----------------------------------------------------------------

extern "C" void kernel_launch(void* const* d_in, const int* in_sizes, int n_in,
                              void* d_out, int out_size, void* d_ws, size_t ws_size,
                              hipStream_t stream) {
  const float* x     = (const float*)d_in[0];
  const int*   eidx  = (const int*)d_in[1];
  const float* eattr = (const float*)d_in[2];
  const float* gamma = (const float*)d_in[3];
  const float* beta  = (const float*)d_in[4];
  const float* w_in1 = (const float*)d_in[5];
  const float* b_in1 = (const float*)d_in[6];
  const float* w_in2 = (const float*)d_in[7];
  const float* b_in2 = (const float*)d_in[8];
  const float* w_c1  = (const float*)d_in[9];
  const float* b_c1  = (const float*)d_in[10];
  const float* w_c2  = (const float*)d_in[11];
  const float* b_c2  = (const float*)d_in[12];
  const float* w_e1  = (const float*)d_in[13];
  const float* b_e1  = (const float*)d_in[14];
  const float* w_e2  = (const float*)d_in[15];
  const float* b_e2  = (const float*)d_in[16];

  const int N = in_sizes[0] / 3;
  const int E = in_sizes[1] / 2;

  char*  ws    = (char*)d_ws;
  float* stats = (float*)ws;                                        // 256 B
  float* feat  = (float*)(ws + 256);                                // N*36 f32
  float* Hn    = (float*)(ws + 256 + (size_t)N * 36 * 4);           // N*32 f32
  float* feat2 = (float*)(ws + 256 + (size_t)N * 36 * 4 + (size_t)N * 32 * 4);
  float* out   = (float*)d_out;

  // Zero scratch that is accumulated into (capture-safe, no memset API).
  zero_kernel<<<64, 256, 0, stream>>>(stats, 64);
  zero_kernel<<<2048, 256, 0, stream>>>(Hn, (long)N * 32);

  bn_stats_kernel<<<256, 256, 0, stream>>>(x, stats, N);
  node_kernel<<<(N + 255) / 256, 256, 0, stream>>>(x, gamma, beta, w_in1, b_in1,
                                                   w_in2, b_in2, stats, feat, N);

  const int numTiles = (E + 15) / 16;
  int grid = numTiles < 8192 ? numTiles : 8192;
  conv_kernel<<<grid, 32, 0, stream>>>(feat, eidx, eattr, w_c1, b_c1, w_c2, b_c2,
                                       Hn, E, numTiles);

  feat2_kernel<<<(int)(((long)N * 36 + 255) / 256), 256, 0, stream>>>(Hn, feat, feat2, N);

  edgeout_kernel<<<grid, 32, 0, stream>>>(feat2, eidx, eattr, w_e1, b_e1, w_e2,
                                          b_e2, out, E, numTiles);
}
